// SPWLoss_1168231105198
// MI455X (gfx1250) — matrix operand order repaired and checked by reading, not compiled, not run on hardware
//
#include <hip/hip_runtime.h>

// ---------------------------------------------------------------------------
// SPW boundary-weighted CE loss for MI455X (gfx1250, wave32).
// 3D box filters (k=3,5,7) computed as banded-matrix WMMAs (f16 exact for
// 0/1 data, sums <= 343), separable x/y/z passes, then fused NLL+reduction.
// ---------------------------------------------------------------------------

typedef _Float16 v16h __attribute__((ext_vector_type(16)));
typedef _Float16 v8h  __attribute__((ext_vector_type(8)));
typedef float    v8f  __attribute__((ext_vector_type(8)));

#define Wd 192
#define Hd 192
#define Dd 96
#define Bd 2
#define DHWc 3538944ull          // 96*192*192
#define NVOX 7077888ull          // 2*DHW
#define NB4  2048                // pass4 blocks

// A-operand (16x32 f16) element i of lane l lives at K:
// lanes 0-15:  K 0..15
// lanes 16-31: K 16..31
__device__ __forceinline__ int a_kidx(int lane, int i) {
  return i + ((lane & 16) ? 16 : 0);
}

// Banded 0/1 matrix as A-operand: rows M = output offset, window covers
// input K in [M+3-rad, M+3+rad]  (inputs start at out_base - 3).
__device__ __forceinline__ v16h band_a(int lane, int rad) {
  v16h a;
  int m = lane & 15;
#pragma unroll
  for (int i = 0; i < 16; ++i) {
    int K = a_kidx(lane, i);
    a[i] = (K >= m + 3 - rad && K <= m + 3 + rad) ? (_Float16)1.0f : (_Float16)0.0f;
  }
  return a;
}

// Banded 0/1 matrix as B-operand: columns N = output offset.
// B layout: lane l -> column N = l&15, K = (l<16 ? 0 : 16) + i.
__device__ __forceinline__ v16h band_b(int lane, int rad) {
  v16h b;
  int n = lane & 15;
  int kb = (lane & 16) ? 16 : 0;
#pragma unroll
  for (int i = 0; i < 16; ++i) {
    int K = kb + i;
    b[i] = (K >= n + 3 - rad && K <= n + 3 + rad) ? (_Float16)1.0f : (_Float16)0.0f;
  }
  return b;
}

__device__ __forceinline__ v8f wmma_f16(v16h a, v16h b, v8f c) {
  return __builtin_amdgcn_wmma_f32_16x16x32_f16(false, a, false, b, (short)0, c,
                                                false, false);
}

__global__ void k_init(int* gmax) { *gmax = 0; }

// ---- Pass 1: window sums along X.  D(16x16) = Band(16x32) x Data(32x16).
// M = x_out, K = x halo, N = flattened (b,z,y) line.  27648 tiles, 8 waves/WG.
__global__ void __launch_bounds__(256)
k_pass1(const int* __restrict__ labels, _Float16* __restrict__ s0,
        _Float16* __restrict__ s1, _Float16* __restrict__ s2) {
  int lane = threadIdx.x & 31;
  int tile = blockIdx.x * 8 + (threadIdx.x >> 5);
  int xbase = (tile % 12) * 16;
  size_t line0 = (size_t)(tile / 12) * 16;

  int n = lane & 15;                       // B column = line offset
  int kb = (lane & 16) ? 16 : 0;
  const int* row = labels + (line0 + n) * Wd;
  v16h bdat;
#pragma unroll
  for (int i = 0; i < 16; ++i) {
    int x = xbase - 3 + kb + i;
    int v = (x >= 0 && x < Wd) ? row[x] : 0;
    bdat[i] = (_Float16)(float)v;
  }
  v8f z = {};
  v8f c0 = wmma_f16(band_a(lane, 1), bdat, z);
  v8f c1 = wmma_f16(band_a(lane, 2), bdat, z);
  v8f c2 = wmma_f16(band_a(lane, 3), bdat, z);

  int mb = (lane & 16) ? 8 : 0;            // D: lane holds M = mb+r, N = lane&15
  size_t off = (line0 + n) * Wd + xbase + mb;
  v8h o0, o1, o2;
#pragma unroll
  for (int r = 0; r < 8; ++r) {
    o0[r] = (_Float16)c0[r]; o1[r] = (_Float16)c1[r]; o2[r] = (_Float16)c2[r];
  }
  *(v8h*)(s0 + off) = o0; *(v8h*)(s1 + off) = o1; *(v8h*)(s2 + off) = o2;
}

// ---- Pass 2: window sums along Y.  D(16x16) = Data(16x32) x Band(32x16).
// M = x, K = y halo, N = y_out.  Fixed (b,z) plane per tile.
__global__ void __launch_bounds__(256)
k_pass2(const _Float16* __restrict__ s0, const _Float16* __restrict__ s1,
        const _Float16* __restrict__ s2, _Float16* __restrict__ t0,
        _Float16* __restrict__ t1, _Float16* __restrict__ t2) {
  int lane = threadIdx.x & 31;
  int tile = blockIdx.x * 8 + (threadIdx.x >> 5);
  int xbase = (tile % 12) * 16;
  int ybase = ((tile / 12) % 12) * 16;
  int bz = tile / 144;                     // b*Dd + z, 0..191
  size_t plane = (size_t)bz * Hd * Wd;
  int x = xbase + (lane & 15);             // A row = x offset

  v16h d0, d1, d2;
#pragma unroll
  for (int i = 0; i < 16; ++i) {
    int y = ybase - 3 + a_kidx(lane, i);
    bool ok = (y >= 0) && (y < Hd);
    size_t idx = plane + (size_t)(ok ? y : 0) * Wd + x;
    d0[i] = ok ? s0[idx] : (_Float16)0.0f;
    d1[i] = ok ? s1[idx] : (_Float16)0.0f;
    d2[i] = ok ? s2[idx] : (_Float16)0.0f;
  }
  v8f z = {};
  v8f c0 = wmma_f16(d0, band_b(lane, 1), z);
  v8f c1 = wmma_f16(d1, band_b(lane, 2), z);
  v8f c2 = wmma_f16(d2, band_b(lane, 3), z);

  int mb = (lane & 16) ? 8 : 0;
  size_t off = plane + (size_t)(ybase + (lane & 15)) * Wd + xbase + mb;
  v8h o0, o1, o2;
#pragma unroll
  for (int r = 0; r < 8; ++r) {
    o0[r] = (_Float16)c0[r]; o1[r] = (_Float16)c1[r]; o2[r] = (_Float16)c2[r];
  }
  *(v8h*)(t0 + off) = o0; *(v8h*)(t1 + off) = o1; *(v8h*)(t2 + off) = o2;
}

// ---- Pass 3: window sums along Z + weight-map + global max.
// M = x, K = z halo, N = z_out.  Fixed (b,y) per tile.
__global__ void __launch_bounds__(256)
k_pass3(const _Float16* __restrict__ t0, const _Float16* __restrict__ t1,
        const _Float16* __restrict__ t2, const int* __restrict__ labels,
        float* __restrict__ wmap, int* __restrict__ gmax) {
  int lane = threadIdx.x & 31;
  int tile = blockIdx.x * 8 + (threadIdx.x >> 5);
  int xbase = (tile % 12) * 16;
  int zbase = ((tile / 12) % 6) * 16;
  int by = tile / 72;                      // 0..383
  int b = by / Hd, y = by % Hd;
  int x0 = xbase + (lane & 15);

  v16h d0, d1, d2;
#pragma unroll
  for (int i = 0; i < 16; ++i) {
    int zc = zbase - 3 + a_kidx(lane, i);
    bool ok = (zc >= 0) && (zc < Dd);
    size_t idx = ((size_t)(b * Dd + (ok ? zc : 0)) * Hd + y) * Wd + x0;
    d0[i] = ok ? t0[idx] : (_Float16)0.0f;
    d1[i] = ok ? t1[idx] : (_Float16)0.0f;
    d2[i] = ok ? t2[idx] : (_Float16)0.0f;
  }
  v8f z = {};
  v8f c0 = wmma_f16(d0, band_b(lane, 1), z);   // S for k=3
  v8f c1 = wmma_f16(d1, band_b(lane, 2), z);   // k=5
  v8f c2 = wmma_f16(d2, band_b(lane, 3), z);   // k=7

  int zo = zbase + (lane & 15);
  int mb = (lane & 16) ? 8 : 0;
  size_t off = ((size_t)(b * Dd + zo) * Hd + y) * Wd + xbase + mb;
  const int* lr = labels + off;

  float lmax = 0.0f;
  v8f wv;
#pragma unroll
  for (int r = 0; r < 8; ++r) {
    int xx = xbase + mb + r;
    float mlab = (float)lr[r];               // onehot channel 1
    float S[3] = {c0[r], c1[r], c2[r]};
    float w = 0.0f;
#pragma unroll
    for (int s = 0; s < 3; ++s) {
      int rad = s + 1, k = 2 * rad + 1;
      float inv_k3 = 1.0f / (float)(k * k * k);
      int nx = min(xx + rad, Wd - 1) - max(xx - rad, 0) + 1;
      int ny = min(y  + rad, Hd - 1) - max(y  - rad, 0) + 1;
      int nz = min(zo + rad, Dd - 1) - max(zo - rad, 0) + 1;
      float nin = (float)(nx * ny * nz);
      w += fabsf(mlab - S[s] * inv_k3) +
           fabsf((1.0f - mlab) - (nin - S[s]) * inv_k3);
    }
    wv[r] = w;
    lmax = fmaxf(lmax, w);
  }
  *(v8f*)(wmap + off) = wv;

  // wave32 max reduce, then one atomic per wave (int order == float order, w>=0)
#pragma unroll
  for (int o = 16; o > 0; o >>= 1) lmax = fmaxf(lmax, __shfl_xor(lmax, o, 32));
  if (lane == 0) atomicMax(gmax, __float_as_int(lmax));
}

// ---- Pass 4: weighted NLL, deterministic block partial sums.
__global__ void __launch_bounds__(256)
k_pass4(const float* __restrict__ logits, const int* __restrict__ labels,
        const float* __restrict__ wmap, const int* __restrict__ gmax,
        float* __restrict__ partials) {
  __shared__ float red[256];
  float g = __int_as_float(*gmax);
  float scale = (g > 0.0f) ? (2.0f / g) : 0.0f;  // BOUNDARY_WEIGHT folded in
  float acc = 0.0f;
  for (size_t i = (size_t)blockIdx.x * 256 + threadIdx.x; i < NVOX;
       i += (size_t)NB4 * 256) {
    size_t b = i / DHWc, rem = i - b * DHWc;
    const float* lp = logits + b * 2 * DHWc + rem;
    float l0 = lp[0], l1 = lp[DHWc];
    int lab = labels[i];
    float mx = fmaxf(l0, l1);
    float lse = mx + __logf(__expf(l0 - mx) + __expf(l1 - mx));
    float nll = lse - (lab ? l1 : l0);
    acc += nll * (1.0f + wmap[i] * scale);
  }
  red[threadIdx.x] = acc;
  __syncthreads();
  for (int st = 128; st > 0; st >>= 1) {
    if (threadIdx.x < (unsigned)st) red[threadIdx.x] += red[threadIdx.x + st];
    __syncthreads();
  }
  if (threadIdx.x == 0) partials[blockIdx.x] = red[0];
}

__global__ void __launch_bounds__(256)
k_final(const float* __restrict__ partials, float* __restrict__ out) {
  __shared__ float red[256];
  float a = 0.0f;
  for (int i = threadIdx.x; i < NB4; i += 256) a += partials[i];
  red[threadIdx.x] = a;
  __syncthreads();
  for (int st = 128; st > 0; st >>= 1) {
    if (threadIdx.x < (unsigned)st) red[threadIdx.x] += red[threadIdx.x + st];
    __syncthreads();
  }
  if (threadIdx.x == 0) out[0] = red[0] / (float)NVOX;
}

extern "C" void kernel_launch(void* const* d_in, const int* in_sizes, int n_in,
                              void* d_out, int out_size, void* d_ws, size_t ws_size,
                              hipStream_t stream) {
  (void)in_sizes; (void)n_in; (void)out_size; (void)ws_size;
  const float* logits = (const float*)d_in[0];
  const int*   labels = (const int*)d_in[1];
  float* out = (float*)d_out;

  char* ws = (char*)d_ws;
  size_t o = 0;
  auto al = [](size_t v) { return (v + 255) & ~(size_t)255; };
  _Float16* sx0 = (_Float16*)(ws + o); o = al(o + NVOX * 2);
  _Float16* sx1 = (_Float16*)(ws + o); o = al(o + NVOX * 2);
  _Float16* sx2 = (_Float16*)(ws + o); o = al(o + NVOX * 2);
  _Float16* sy0 = (_Float16*)(ws + o); o = al(o + NVOX * 2);
  _Float16* sy1 = (_Float16*)(ws + o); o = al(o + NVOX * 2);
  _Float16* sy2 = (_Float16*)(ws + o); o = al(o + NVOX * 2);
  float*    wmap = (float*)(ws + o);   o = al(o + NVOX * 4);
  int*      gmax = (int*)(ws + o);     o = al(o + 4);
  float*    partials = (float*)(ws + o);

  // 27648 tiles per stencil pass, 8 waves (one tile each) per 256-thread block.
  k_init <<<1, 1, 0, stream>>>(gmax);
  k_pass1<<<3456, 256, 0, stream>>>(labels, sx0, sx1, sx2);
  k_pass2<<<3456, 256, 0, stream>>>(sx0, sx1, sx2, sy0, sy1, sy2);
  k_pass3<<<3456, 256, 0, stream>>>(sy0, sy1, sy2, labels, wmap, gmax);
  k_pass4<<<NB4, 256, 0, stream>>>(logits, labels, wmap, gmax, partials);
  k_final<<<1, 256, 0, stream>>>(partials, out);
}
